// IASA_2018634629361
// MI455X (gfx1250) — matrix operand order repaired and checked by reading, not compiled, not run on hardware
//
#include <hip/hip_runtime.h>
#include <hip/hip_bf16.h>
#include <math.h>

// ---------------------------------------------------------------------------
// MI455X (gfx1250) IASA forward.
// ~107 GFLOP; f16-WMMA-bound (~21us at ~5 PFLOPS dense f16) vs ~7us mandatory
// HBM at 23.3 TB/s. All matrix math on v_wmma_f32_16x16x32_f16, f16
// intermediates (~130MB -> resident in the 192MB L2), gather fused into the
// QKV GEMM, scatter fused into the attention epilogue, flash softmax.
// This round: x / attn are staged in LDS ONCE per block and all column tiles
// are produced from that staging (kills the 24x / 8x redundant HBM re-reads),
// and K-window + proj A staging use global_load_async_to_lds_b128 (ASYNCcnt).
// ---------------------------------------------------------------------------

typedef __attribute__((ext_vector_type(16))) _Float16 v16h;
typedef __attribute__((ext_vector_type(8)))  float    v8f;

#define DIM      512
#define HEADS    8
#define HD       64      // head dim (qk and v)
#define GS       256     // group size
#define WIN      512     // local window = 2*GS
#define MG       64      // global memory slots
#define NEG_BIG  (-1e30f)

__device__ __forceinline__ v8f wmma_f16(v16h a, v16h b, v8f c) {
  return __builtin_amdgcn_wmma_f32_16x16x32_f16(
      /*neg_a=*/false, a, /*neg_b=*/false, b,
      /*c_mod=*/(short)0, c, /*reuse_a=*/false, /*reuse_b=*/false);
}

// Async DMA: 16B global -> LDS, tracked by ASYNCcnt (no VGPR round trip).
// lds_off is the byte offset inside the workgroup's LDS allocation.
__device__ __forceinline__ void async_ld_b128(unsigned lds_off, const void* gsrc) {
  asm volatile("global_load_async_to_lds_b128 %0, %1, off"
               :: "v"(lds_off), "v"(gsrc) : "memory");
}

// A fragment 16x32 f16 (ISA 7.12.2): lane m=lane&15, K interleaved per half:
// half 0 -> K in {0..7,16..23}, half 1 -> K in {8..15,24..31}; lowers to
// 2x ds_load_b128 / global_load_b128 per lane.
__device__ __forceinline__ v16h load_a_frag(const _Float16* base, int stride, int kbase) {
  const int lane  = threadIdx.x & 31;
  const int m     = lane & 15;
  const int khalf = (lane >> 4) * 8;
  const _Float16* row = base + (size_t)m * stride + kbase;
  v16h a;
#pragma unroll
  for (int i = 0; i < 8; ++i) {
    const int K = ((i < 4) ? 0 : 16) + khalf + 2 * (i & 3);
    a[2 * i]     = row[K];
    a[2 * i + 1] = row[K + 1];
  }
  return a;
}

// B fragment 32x16 f16: lane n=lane&15; K interleaved per half like A:
// half 0 -> K in {0..7,16..23}, half 1 -> K in {8..15,24..31}. Element (K,n) read from tbase[n*stride + kbase + K].
__device__ __forceinline__ v16h load_b_frag(const _Float16* tbase, int stride, int kbase) {
  const int lane = threadIdx.x & 31;
  const int n    = lane & 15;
  const int koff = (lane >> 4) * 8;
  const _Float16* col = tbase + (size_t)n * stride + kbase + koff;
  v16h b;
#pragma unroll
  for (int i = 0; i < 8; ++i) {
    b[i]     = col[i];
    b[i + 8] = col[16 + i];
  }
  return b;
}

// ---------------------------------------------------------------------------
// Kernel 1: weight f32 [k][n] -> f16 transposed [n][k] (B-operand friendly).
// ---------------------------------------------------------------------------
__global__ void wcvt_kernel(const float* __restrict__ W, _Float16* __restrict__ Wt) {
  int i = blockIdx.x * 256 + threadIdx.x;       // 512*512 elements
  int k = i >> 9, n = i & 511;
  Wt[(size_t)n * DIM + k] = (_Float16)W[i];
}

// ---------------------------------------------------------------------------
// Kernel 2: fused gather + QKV GEMM.  grid = (B*N/128), block 256 (8 waves).
// Each wave stages its 16 gathered x-rows in LDS ONCE (f32->f16), then emits
// all 24 output tiles (q,k,v x 8 col-tiles): x is read from HBM exactly once.
// Per wave: 24 * 16 * 4 = 1536 WMMAs.
// ---------------------------------------------------------------------------
__global__ void qkv_kernel(const float* __restrict__ x, const int* __restrict__ idx,
                           const _Float16* __restrict__ WqT, const _Float16* __restrict__ WkT,
                           const _Float16* __restrict__ WvT,
                           _Float16* __restrict__ qh, _Float16* __restrict__ kh,
                           _Float16* __restrict__ vh, int N) {
  extern __shared__ char smem[];
  _Float16* sA = (_Float16*)smem;               // 8 waves * 16 * 512 f16 = 128KB
  const int tid = threadIdx.x, wave = tid >> 5, lane = tid & 31;
  const int globalRow0 = blockIdx.x * 128 + wave * 16;   // over B*N, batch-aligned
  const int b = globalRow0 / N;
  const int rowInB = globalRow0 - b * N;
  _Float16* sAw = sA + (size_t)wave * 16 * DIM;

  { // stage 16 gathered rows, f32 -> f16 (lane r = lane>>1, half-row each)
    const int r  = lane >> 1;
    const int c0 = (lane & 1) * 256;
    const int src = idx[(size_t)b * N + rowInB + r];
    const float* xr = x + ((size_t)b * N + src) * DIM + c0;
    _Float16* dr = sAw + (size_t)r * DIM + c0;
    __builtin_prefetch(xr, 0, 0);               // global_prefetch_b8
#pragma unroll 4
    for (int c = 0; c < 256; c += 4) {
      float4 f = *(const float4*)(xr + c);
      dr[c + 0] = (_Float16)f.x;  dr[c + 1] = (_Float16)f.y;
      dr[c + 2] = (_Float16)f.z;  dr[c + 3] = (_Float16)f.w;
    }
  }
  asm volatile("s_wait_dscnt 0" ::: "memory");  // wave-local LDS RAW

  const int nI = lane & 15, mH = (lane >> 4) * 8;
  for (int sel = 0; sel < 3; ++sel) {
    const _Float16* WT = (sel == 0) ? WqT : (sel == 1) ? WkT : WvT;
    _Float16* dst; size_t base;
    if (sel == 0) { dst = qh; base = ((size_t)b * N + rowInB) * DIM; }
    else { dst = (sel == 1) ? kh : vh; base = ((size_t)b * (N + GS) + rowInB) * DIM; }
    for (int colBase = 0; colBase < DIM; colBase += 64) {
      v8f acc0 = {}, acc1 = {}, acc2 = {}, acc3 = {};
#pragma unroll 4
      for (int kc = 0; kc < 16; ++kc) {
        v16h a  = load_a_frag(sAw, DIM, kc * 32);
        v16h b0 = load_b_frag(WT + (size_t)(colBase +  0) * DIM, DIM, kc * 32);
        v16h b1 = load_b_frag(WT + (size_t)(colBase + 16) * DIM, DIM, kc * 32);
        v16h b2 = load_b_frag(WT + (size_t)(colBase + 32) * DIM, DIM, kc * 32);
        v16h b3 = load_b_frag(WT + (size_t)(colBase + 48) * DIM, DIM, kc * 32);
        acc0 = wmma_f16(a, b0, acc0);  acc1 = wmma_f16(a, b1, acc1);
        acc2 = wmma_f16(a, b2, acc2);  acc3 = wmma_f16(a, b3, acc3);
      }
      v8f* accs[4] = {&acc0, &acc1, &acc2, &acc3};
#pragma unroll
      for (int t = 0; t < 4; ++t)
#pragma unroll
        for (int r = 0; r < 8; ++r)
          dst[base + (size_t)(mH + r) * DIM + colBase + t * 16 + nI] =
              (_Float16)((*accs[t])[r]);
    }
  }
}

// ---------------------------------------------------------------------------
// Kernel 3: mirror-pad K/V rows N..N+255 with flipped tail (pad_n == 0 case).
// ---------------------------------------------------------------------------
__global__ void mirror_kernel(_Float16* __restrict__ kh, _Float16* __restrict__ vh, int N) {
  int e = blockIdx.x * 256 + threadIdx.x;       // B*256*512
  int b = e / (GS * DIM);
  int rem = e - b * GS * DIM;
  int t = rem >> 9, c = rem & 511;
  size_t batch = (size_t)b * (N + GS) * DIM;
  kh[batch + (size_t)(N + t) * DIM + c] = kh[batch + (size_t)(N - 1 - t) * DIM + c];
  vh[batch + (size_t)(N + t) * DIM + c] = vh[batch + (size_t)(N - 1 - t) * DIM + c];
}

// ---------------------------------------------------------------------------
// Flash online-softmax over nkeys (multiple of 32) for one 16-row query tile.
// sK: [key][kstride] row-major (K^T B-frags), sVt: [dim][vstride] transposed.
// P converted D-layout -> A-layout through wave-local LDS scratch sP (16x32).
// ---------------------------------------------------------------------------
struct FlashState { v8f o[4]; float m[8]; float l[8]; };

__device__ __forceinline__ void flash_init(FlashState& st) {
#pragma unroll
  for (int t = 0; t < 4; ++t) st.o[t] = (v8f){};
#pragma unroll
  for (int r = 0; r < 8; ++r) { st.m[r] = NEG_BIG; st.l[r] = 0.0f; }
}

__device__ __forceinline__ void flash_run(const v16h aq[2],
                                          const _Float16* sK, int kstride,
                                          const _Float16* sVt, int vstride,
                                          int nkeys, float scale,
                                          _Float16* sP, FlashState& st) {
  const int lane = threadIdx.x & 31;
  const int nIdx = lane & 15;
  const int mHi  = (lane >> 4) * 8;
  for (int kb = 0; kb < nkeys; kb += 32) {
    v8f s0 = {}, s1 = {};
#pragma unroll
    for (int kc = 0; kc < 2; ++kc) {
      v16h bk0 = load_b_frag(sK + (size_t)kb * kstride,        kstride, kc * 32);
      v16h bk1 = load_b_frag(sK + (size_t)(kb + 16) * kstride, kstride, kc * 32);
      s0 = wmma_f16(aq[kc], bk0, s0);
      s1 = wmma_f16(aq[kc], bk1, s1);
    }
    // online softmax update, row stats replicated across each 16-lane half
#pragma unroll
    for (int r = 0; r < 8; ++r) {
      float a0 = s0[r] * scale, a1 = s1[r] * scale;
      float mx = fmaxf(a0, a1);
#pragma unroll
      for (int w = 1; w < 16; w <<= 1) mx = fmaxf(mx, __shfl_xor(mx, w, 16));
      float mnew = fmaxf(st.m[r], mx);
      float corr = __expf(st.m[r] - mnew);
      float p0 = __expf(a0 - mnew), p1 = __expf(a1 - mnew);
      float rs = p0 + p1;
#pragma unroll
      for (int w = 1; w < 16; w <<= 1) rs += __shfl_xor(rs, w, 16);
      st.l[r] = st.l[r] * corr + rs;
      st.m[r] = mnew;
#pragma unroll
      for (int t = 0; t < 4; ++t) st.o[t][r] *= corr;
      // stash P (D-layout) into LDS for A-layout reload
      sP[(size_t)(r + mHi) * 32 + nIdx]      = (_Float16)p0;
      sP[(size_t)(r + mHi) * 32 + 16 + nIdx] = (_Float16)p1;
    }
    asm volatile("s_wait_dscnt 0" ::: "memory");
    v16h ap = load_a_frag(sP, 32, 0);
#pragma unroll
    for (int t = 0; t < 4; ++t) {
      v16h bv = load_b_frag(sVt + (size_t)(t * 16) * vstride + kb, vstride, 0);
      st.o[t] = wmma_f16(ap, bv, st.o[t]);
    }
    asm volatile("s_wait_dscnt 0" ::: "memory");  // before sP rewrite next chunk
  }
}

// ---------------------------------------------------------------------------
// Kernel 4: attention. grid = (ng, H, B), block 256 (8 waves x 32 query rows).
// K window staged by async DMA (global_load_async_to_lds_b128, ASYNCcnt);
// V window transposed in LDS for wide B-fragment reads. 152KB LDS of 320KB.
// ---------------------------------------------------------------------------
__global__ void attn_kernel(const _Float16* __restrict__ qh, const _Float16* __restrict__ kh,
                            const _Float16* __restrict__ vh,
                            const float* __restrict__ kg, const float* __restrict__ vg,
                            const int* __restrict__ idx, _Float16* __restrict__ aout, int N) {
  const int g = blockIdx.x, h = blockIdx.y, b = blockIdx.z;
  extern __shared__ char smem[];
  _Float16* sK   = (_Float16*)smem;          // [512][64]  (LDS offset 0)
  _Float16* sVt  = sK  + WIN * HD;           // [64][512]
  _Float16* sKg  = sVt + HD * WIN;           // [64][64]
  _Float16* sVgt = sKg + MG * HD;            // [64][64]
  _Float16* sP   = sVgt + HD * MG;           // 8 waves * 16 * 32
  const int tid = threadIdx.x;
  const size_t kvBatch = (size_t)(N + GS) * DIM;
  const size_t winBase = (size_t)b * kvBatch + (size_t)(g * GS) * DIM + h * HD;

  // async-DMA the local K window: 512 rows x 128B = 4096 x 16B transfers
  for (int t = tid; t < WIN * 8; t += 256) {
    const int kr = t >> 3, ch = (t & 7) * 8;       // 8 halves per 16B chunk
    async_ld_b128((unsigned)((kr * HD + ch) * sizeof(_Float16)),
                  kh + winBase + (size_t)kr * DIM + ch);
  }
  // stage V transposed (one-time scalar stores, amortized by wide frag reads)
  for (int kr = tid; kr < WIN; kr += 256) {
    const _Float16* vrow = vh + winBase + (size_t)kr * DIM;
#pragma unroll 8
    for (int d = 0; d < HD; ++d) sVt[(size_t)d * WIN + kr] = vrow[d];
  }
  // stage global K/V (f32 -> f16)
  for (int e = tid; e < MG * HD; e += 256) {
    int mm = e >> 6, d = e & 63;
    sKg[(size_t)mm * HD + d]  = (_Float16)kg[((size_t)h * MG + mm) * HD + d];
    sVgt[(size_t)d * MG + mm] = (_Float16)vg[((size_t)h * MG + mm) * HD + d];
  }
  asm volatile("s_wait_asynccnt 0" ::: "memory");
  __syncthreads();

  const int wave = tid >> 5, lane = tid & 31;
  const int nIdx = lane & 15, mHi = (lane >> 4) * 8;
  const float scale = 0.125f;                // 1/sqrt(64)
  _Float16* sPw = sP + (size_t)wave * 16 * 32;

  for (int rt = 0; rt < 2; ++rt) {
    const int qrow0 = wave * 32 + rt * 16;   // within group
    const _Float16* qbase = qh + ((size_t)b * N + (size_t)g * GS + qrow0) * DIM + h * HD;
    v16h aq[2];
    aq[0] = load_a_frag(qbase, DIM, 0);
    aq[1] = load_a_frag(qbase, DIM, 32);

    FlashState st;
    flash_init(st);
    flash_run(aq, sK, HD, sVt, WIN, WIN, scale, sPw, st);   // local window
    float o1[4][8];
#pragma unroll
    for (int t = 0; t < 4; ++t)
#pragma unroll
      for (int r = 0; r < 8; ++r) o1[t][r] = st.o[t][r] / st.l[r];

    flash_init(st);
    flash_run(aq, sKg, HD, sVgt, MG, MG, scale, sPw, st);   // global slots

    // combine, scatter back through idx_last (inverse of the gather)
#pragma unroll
    for (int r = 0; r < 8; ++r) {
      const int tok = g * GS + qrow0 + mHi + r;
      const int dstRow = idx[(size_t)b * N + tok];
      _Float16* drow = aout + ((size_t)b * N + dstRow) * DIM + h * HD;
#pragma unroll
      for (int t = 0; t < 4; ++t)
        drow[t * 16 + nIdx] = (_Float16)(o1[t][r] + st.o[t][r] / st.l[r]);
    }
  }
}

// ---------------------------------------------------------------------------
// Kernel 5: projection GEMM, f16 x f16 -> f32. grid = (B*N/128), block 256.
// A rows async-DMA'd to LDS once per wave, then all 8 column tiles produced.
// ---------------------------------------------------------------------------
__global__ void proj_kernel(const _Float16* __restrict__ ain, const _Float16* __restrict__ WpT,
                            float* __restrict__ out) {
  extern __shared__ char smem[];
  _Float16* sA = (_Float16*)smem;               // 8 waves * 16 * 512 = 128KB
  const int tid = threadIdx.x, wave = tid >> 5, lane = tid & 31;
  const int row0 = blockIdx.x * 128 + wave * 16;
  const _Float16* abase = ain + (size_t)row0 * DIM;
  const unsigned waveOff = (unsigned)(wave * 16 * DIM * sizeof(_Float16));

  // async-DMA 16 rows x 1KB = 1024 x 16B transfers per wave (32 per lane)
  for (int t = lane; t < 16 * (DIM / 8); t += 32) {
    const int row = t >> 6, ch = (t & 63) * 8;
    async_ld_b128(waveOff + (unsigned)((row * DIM + ch) * sizeof(_Float16)),
                  abase + (size_t)row * DIM + ch);
  }
  asm volatile("s_wait_asynccnt 0" ::: "memory");  // wave-local staging

  _Float16* sAw = sA + (size_t)wave * 16 * DIM;
  const int nI = lane & 15, mH = (lane >> 4) * 8;
  for (int colBase = 0; colBase < DIM; colBase += 64) {
    v8f acc0 = {}, acc1 = {}, acc2 = {}, acc3 = {};
#pragma unroll 4
    for (int kc = 0; kc < 16; ++kc) {
      v16h a  = load_a_frag(sAw, DIM, kc * 32);
      v16h b0 = load_b_frag(WpT + (size_t)(colBase +  0) * DIM, DIM, kc * 32);
      v16h b1 = load_b_frag(WpT + (size_t)(colBase + 16) * DIM, DIM, kc * 32);
      v16h b2 = load_b_frag(WpT + (size_t)(colBase + 32) * DIM, DIM, kc * 32);
      v16h b3 = load_b_frag(WpT + (size_t)(colBase + 48) * DIM, DIM, kc * 32);
      acc0 = wmma_f16(a, b0, acc0);  acc1 = wmma_f16(a, b1, acc1);
      acc2 = wmma_f16(a, b2, acc2);  acc3 = wmma_f16(a, b3, acc3);
    }
    v8f* accs[4] = {&acc0, &acc1, &acc2, &acc3};
#pragma unroll
    for (int t = 0; t < 4; ++t)
#pragma unroll
      for (int r = 0; r < 8; ++r)
        out[(size_t)(row0 + mH + r) * DIM + colBase + t * 16 + nI] = (*accs[t])[r];
  }
}

// ---------------------------------------------------------------------------
extern "C" void kernel_launch(void* const* d_in, const int* in_sizes, int n_in,
                              void* d_out, int out_size, void* d_ws, size_t ws_size,
                              hipStream_t stream) {
  const float* x   = (const float*)d_in[0];
  const int*   idx = (const int*)d_in[1];
  const float* kg  = (const float*)d_in[2];
  const float* vg  = (const float*)d_in[3];
  const float* Wq  = (const float*)d_in[4];
  const float* Wk  = (const float*)d_in[5];
  const float* Wv  = (const float*)d_in[6];
  const float* Wp  = (const float*)d_in[7];
  float* out = (float*)d_out;

  const int BN = in_sizes[1];        // idx_last flat count = B*N
  const int B  = 2;
  const int N  = BN / B;
  const int ng = N / GS;

  // workspace layout (f16 intermediates, ~131MB, L2-resident)
  const size_t W_E = (size_t)DIM * DIM;
  _Float16* p    = (_Float16*)d_ws;
  _Float16* WqT  = p; p += W_E;
  _Float16* WkT  = p; p += W_E;
  _Float16* WvT  = p; p += W_E;
  _Float16* WpT  = p; p += W_E;
  _Float16* qh   = p; p += (size_t)B * N * DIM;
  _Float16* kh   = p; p += (size_t)B * (N + GS) * DIM;
  _Float16* vh   = p; p += (size_t)B * (N + GS) * DIM;
  _Float16* aobf = p; p += (size_t)B * N * DIM;

  const int wcvtBlocks = (int)(W_E / 256);
  wcvt_kernel<<<wcvtBlocks, 256, 0, stream>>>(Wq, WqT);
  wcvt_kernel<<<wcvtBlocks, 256, 0, stream>>>(Wk, WkT);
  wcvt_kernel<<<wcvtBlocks, 256, 0, stream>>>(Wv, WvT);
  wcvt_kernel<<<wcvtBlocks, 256, 0, stream>>>(Wp, WpT);

  const size_t gemmLds = (size_t)8 * 16 * DIM * sizeof(_Float16);         // 128KB
  qkv_kernel<<<dim3(BN / 128), 256, gemmLds, stream>>>(
      x, idx, WqT, WkT, WvT, qh, kh, vh, N);

  mirror_kernel<<<(B * GS * DIM) / 256, 256, 0, stream>>>(kh, vh, N);

  const size_t attnLds = (size_t)(WIN * HD + HD * WIN + MG * HD + HD * MG +
                                  8 * 16 * 32) * sizeof(_Float16);        // 152KB
  attn_kernel<<<dim3(ng, HEADS, B), 256, attnLds, stream>>>(
      qh, kh, vh, kg, vg, idx, aobf, N);

  proj_kernel<<<dim3(BN / 128), 256, gemmLds, stream>>>(aobf, WpT, out);
}
